// Split_BN_ReLU_DConv_27882927685674
// MI455X (gfx1250) — matrix operand order repaired and checked by reading, not compiled, not run on hardware
//
#include <hip/hip_runtime.h>

typedef __attribute__((ext_vector_type(16))) _Float16 v16h;
typedef __attribute__((ext_vector_type(8)))  _Float16 v8h;
typedef __attribute__((ext_vector_type(8)))  float    v8f;
typedef __attribute__((ext_vector_type(4)))  float    v4f;

union H16 { v16h v; v8h h[2]; };

constexpr int kB   = 4;
constexpr int kC   = 128;        // activation channels (NHWC innermost stride)
constexpr int kH   = 128;
constexpr int kW   = 128;
constexpr int kHW  = kH * kW;
constexpr int kHP  = kH + 2;     // zero-padded spatial dims (halo = 1)
constexpr int kWP  = kW + 2;
constexpr int kOff = 18;         // 2*K*K offset channels

__device__ __forceinline__ int clampi(int v, int lo, int hi) {
    return v < lo ? lo : (v > hi ? hi : v);
}

// ---------------------------------------------------------------------------
// Zero-fill a half buffer in 16B chunks (halo init for padded NHWC buffers).
// ---------------------------------------------------------------------------
__global__ void zero_h_kernel(_Float16* __restrict__ p, int n8)
{
    int i = blockIdx.x * blockDim.x + threadIdx.x;
    if (i < n8) {
        v8h z = {};
        *(v8h*)(p + (size_t)i * 8) = z;
    }
}

__global__ void zero_kernel(float* __restrict__ out, int n)
{
    int i = blockIdx.x * blockDim.x + threadIdx.x;
    if (i < n) out[i] = 0.f;
}

// ---------------------------------------------------------------------------
// Fused double BN+ReLU; NCHW f32 -> zero-padded NHWC f16 for xa and y.
// One thread per pixel: reads lane-coalesced per channel plane, writes 16B
// v8h chunks into the (y+1, x+1) interior of the padded buffers.
// ---------------------------------------------------------------------------
__global__ void bn_pack_kernel(const float* __restrict__ x,
                               const float* __restrict__ g1, const float* __restrict__ b1,
                               const float* __restrict__ m1, const float* __restrict__ v1,
                               const float* __restrict__ g2, const float* __restrict__ b2,
                               const float* __restrict__ m2, const float* __restrict__ v2,
                               _Float16* __restrict__ xa_nhwc,
                               _Float16* __restrict__ y_nhwc)
{
    int i = blockIdx.x * blockDim.x + threadIdx.x;   // pixel index over B*HW
    if (i >= kB * kHW) return;
    int b  = i >> 14;
    int hw = i & (kHW - 1);
    int py = hw >> 7;
    int px = hw & (kW - 1);
    const float* xp = x + (size_t)b * kC * kHW + hw;
    size_t po = (((size_t)b * kHP + py + 1) * kWP + px + 1) * kC;
    _Float16* xap = xa_nhwc + po;
    _Float16* yp  = y_nhwc  + po;
    for (int c8 = 0; c8 < kC; c8 += 8) {
        v8h xa8, y8;
        #pragma unroll
        for (int j = 0; j < 8; ++j) {
            int c = c8 + j;
            float xv = xp[(size_t)c * kHW];
            float s1 = g1[c] * rsqrtf(v1[c] + 1e-5f);
            float xa = fmaxf((xv - m1[c]) * s1 + b1[c], 0.f);
            float s2 = g2[c] * rsqrtf(v2[c] + 1e-5f);
            float yv = fmaxf((xa - m2[c]) * s2 + b2[c], 0.f);
            xa8[j] = (_Float16)xa;
            y8[j]  = (_Float16)yv;
        }
        *(v8h*)(xap + c8) = xa8;
        *(v8h*)(yp + c8)  = y8;
    }
}

// ---------------------------------------------------------------------------
// Pack 3x3 conv weights W[O][Cin][3][3] (f32) into the WMMA B-operand lane
// layout: dst[nt][step][lane][h] = W[n][k], n = nt*16 + lane%16,
// k = step*32 + (h/8)*16 + (lane/16)*8 + h%8, K ordered k = t*Cin + c (tap-major).
// ---------------------------------------------------------------------------
__global__ void pack_wmma_b_kernel(const float* __restrict__ w,
                                   _Float16* __restrict__ dst,
                                   int O, int Cin, int NT)
{
    int steps = (Cin * 9) / 32;
    int total = NT * steps * 32 * 16;
    int i = blockIdx.x * blockDim.x + threadIdx.x;
    if (i >= total) return;
    int h    = i & 15;
    int lane = (i >> 4) & 31;
    int step = (i >> 9) % steps;
    int nt   = (i >> 9) / steps;
    int n = nt * 16 + (lane & 15);
    int k = step * 32 + (h >> 3) * 16 + (lane >> 4) * 8 + (h & 7);
    int t = k / Cin;
    int c = k % Cin;
    float v = 0.f;
    if (n < O) v = w[((size_t)n * Cin + c) * 9 + t];
    dst[i] = (_Float16)v;
}

// ---------------------------------------------------------------------------
// 3x3 conv via implicit im2col + WMMA on the zero-padded NHWC buffer.
// One wave32 per 32-pixel row tile (two 16-row A tiles sharing each B
// fragment -> 4 v_wmma per K step, branchless unconditional loads).
// B loads index straight off the kernarg pointer (integer offsets only) so
// they stay in the global address space (no flat/DScnt coupling).
// ---------------------------------------------------------------------------
template <int CIN, int NT>
__global__ __launch_bounds__(32)
void conv3x3_wmma_kernel(const _Float16* __restrict__ src,    // [B,kHP,kWP,CIN] padded
                         const _Float16* __restrict__ wpack,  // [NT][steps][32][16]
                         const float* __restrict__ bias,
                         float* __restrict__ dst,             // [B,Nout,HW]
                         int Nout)
{
    constexpr int QN    = CIN / 32;
    constexpr int steps = QN * 9;
    int lane = threadIdx.x;
    int pix0 = blockIdx.x * 32;
    int b   = pix0 >> 14;
    int hw0 = pix0 & (kHW - 1);
    int yy  = hw0 >> 7;
    int xx0 = hw0 & (kW - 1);
    int m   = lane & 15;
    int hi  = lane >> 4;

    v8f zero = {};
    v8f acc[2][NT];
    #pragma unroll
    for (int nt = 0; nt < NT; ++nt) {
        acc[0][nt] = zero;
        acc[1][nt] = zero;
    }

    // pixel (yy, xx0+m) in padded coords; tap (dy,dx) in {0,1,2} adds directly
    const _Float16* lanebase =
        src + (((size_t)b * kHP + yy) * kWP + xx0 + m) * CIN + hi * 8;

    int wofs = lane * 16;     // running B offset (halves); +512 per K step
    for (int dy = 0; dy < 3; ++dy) {
        for (int dx = 0; dx < 3; ++dx) {
            const _Float16* p = lanebase + ((size_t)dy * kWP + dx) * CIN;
            #pragma unroll
            for (int q = 0; q < QN; ++q) {
                int co = q * 32;
                H16 a0, a1;
                a0.h[0] = *(const v8h*)(p + co);
                a0.h[1] = *(const v8h*)(p + co + 16);
                a1.h[0] = *(const v8h*)(p + 16 * CIN + co);
                a1.h[1] = *(const v8h*)(p + 16 * CIN + co + 16);
                #pragma unroll
                for (int nt = 0; nt < NT; ++nt) {
                    H16 bm;
                    bm.h[0] = *(const v8h*)(wpack + nt * (steps * 512) + wofs);
                    bm.h[1] = *(const v8h*)(wpack + nt * (steps * 512) + wofs + 8);
                    acc[0][nt] = __builtin_amdgcn_wmma_f32_16x16x32_f16(
                        false, a0.v, false, bm.v, (short)0, acc[0][nt], false, false);
                    acc[1][nt] = __builtin_amdgcn_wmma_f32_16x16x32_f16(
                        false, a1.v, false, bm.v, (short)0, acc[1][nt], false, false);
                }
                wofs += 512;
            }
        }
    }

    // D layout: n = lane%16, rows m = v + 8*(lane/16) -> 8 consecutive floats
    int nlo = lane & 15;
    #pragma unroll
    for (int mt = 0; mt < 2; ++mt) {
        #pragma unroll
        for (int nt = 0; nt < NT; ++nt) {
            int n = nt * 16 + nlo;
            if (n < Nout) {
                float bv = bias[n];
                float* dp = dst + ((size_t)b * Nout + n) * kHW + hw0 + mt * 16 + hi * 8;
                v4f lo4 = __builtin_shufflevector(acc[mt][nt], acc[mt][nt], 0, 1, 2, 3);
                v4f hi4 = __builtin_shufflevector(acc[mt][nt], acc[mt][nt], 4, 5, 6, 7);
                lo4 += bv;
                hi4 += bv;
                *(v4f*)(dp)     = lo4;
                *(v4f*)(dp + 4) = hi4;
            }
        }
    }
}

// ---------------------------------------------------------------------------
// Deformable conv, fully in registers: each lane builds its WMMA A fragment
// by blending 4 corner v8h loads with packed-f16 FMAs (validity folded into
// the f16 corner weights). Reads the padded NHWC buffer (coords shifted +1).
// If accumulate: out[b][n][hw] += probs[b][prob_ch][hw] * (acc + bias)
// else:          out[b][n][hw]  = acc + bias            (n < CO_real)
// ---------------------------------------------------------------------------
template <int CIN, int NT>
__global__ __launch_bounds__(32)
void deform_wmma_kernel(const _Float16* __restrict__ src,   // [B,kHP,kWP,128] padded
                        int c_base,
                        const float* __restrict__ offs,     // [B,18,HW]
                        const _Float16* __restrict__ wpack, // [NT][steps][32][16]
                        const float* __restrict__ bias,
                        float* __restrict__ out, int CO_real,
                        const float* __restrict__ probs, int prob_ch,
                        int accumulate)
{
    constexpr int QN    = CIN / 32;
    constexpr int steps = QN * 9;
    int lane = threadIdx.x;
    int pix0 = blockIdx.x * 16;
    int b   = pix0 >> 14;
    int hw0 = pix0 & (kHW - 1);
    int yy  = hw0 >> 7;
    int xx0 = hw0 & (kW - 1);
    int m   = lane & 15;
    int hi  = lane >> 4;
    int xx  = xx0 + m;

    v8f zero = {};
    v8f acc[NT];
    #pragma unroll
    for (int nt = 0; nt < NT; ++nt) acc[nt] = zero;

    // interior origin of padded image b (+1 row, +1 col), channel slice, K half
    const _Float16* base =
        src + (((size_t)b * kHP + 1) * kWP + 1) * kC + c_base + hi * 8;

    int wofs = lane * 16;     // running B offset (halves); +512 per K step
    int t = 0;
    for (int dy = -1; dy <= 1; ++dy) {
        for (int dx = -1; dx <= 1; ++dx, ++t) {
            // per-(pixel,tap) sampling coordinates
            size_t ob = ((size_t)b * kOff + 2 * t) * kHW + yy * kW + xx;
            float offy = offs[ob];
            float offx = offs[ob + kHW];
            float py = (float)(yy + dy) + offy;
            float px = (float)(xx + dx) + offx;
            float fy = floorf(py), fx = floorf(px);
            int y0 = (int)fy, x0 = (int)fx;
            int y1 = y0 + 1, x1 = x0 + 1;
            float wy = py - fy, wx = px - fx;
            float vy0 = (y0 >= 0 && y0 < kH) ? 1.f : 0.f;
            float vy1 = (y1 >= 0 && y1 < kH) ? 1.f : 0.f;
            float vx0 = (x0 >= 0 && x0 < kW) ? 1.f : 0.f;
            float vx1 = (x1 >= 0 && x1 < kW) ? 1.f : 0.f;
            int cy0 = clampi(y0, 0, kH - 1), cy1 = clampi(y1, 0, kH - 1);
            int cx0 = clampi(x0, 0, kW - 1), cx1 = clampi(x1, 0, kW - 1);
            const _Float16* p00 = base + ((size_t)cy0 * kWP + cx0) * kC;
            const _Float16* p01 = base + ((size_t)cy0 * kWP + cx1) * kC;
            const _Float16* p10 = base + ((size_t)cy1 * kWP + cx0) * kC;
            const _Float16* p11 = base + ((size_t)cy1 * kWP + cx1) * kC;
            _Float16 w00 = (_Float16)((1.f - wy) * (1.f - wx) * vy0 * vx0);
            _Float16 w01 = (_Float16)((1.f - wy) * wx         * vy0 * vx1);
            _Float16 w10 = (_Float16)(wy         * (1.f - wx) * vy1 * vx0);
            _Float16 w11 = (_Float16)(wy         * wx         * vy1 * vx1);
            #pragma unroll
            for (int q = 0; q < QN; ++q) {
                int co = q * 32;
                H16 a;
                a.h[0] = (*(const v8h*)(p00 + co)) * w00
                       + (*(const v8h*)(p01 + co)) * w01
                       + (*(const v8h*)(p10 + co)) * w10
                       + (*(const v8h*)(p11 + co)) * w11;
                a.h[1] = (*(const v8h*)(p00 + co + 16)) * w00
                       + (*(const v8h*)(p01 + co + 16)) * w01
                       + (*(const v8h*)(p10 + co + 16)) * w10
                       + (*(const v8h*)(p11 + co + 16)) * w11;
                #pragma unroll
                for (int nt = 0; nt < NT; ++nt) {
                    H16 bm;
                    bm.h[0] = *(const v8h*)(wpack + nt * (steps * 512) + wofs);
                    bm.h[1] = *(const v8h*)(wpack + nt * (steps * 512) + wofs + 8);
                    acc[nt] = __builtin_amdgcn_wmma_f32_16x16x32_f16(
                        false, a.v, false, bm.v, (short)0, acc[nt], false, false);
                }
                wofs += 512;
            }
        }
    }

    int nlo = lane & 15;
    #pragma unroll
    for (int nt = 0; nt < NT; ++nt) {
        int n = nt * 16 + nlo;
        if (n < CO_real) {
            float bv = bias[n];
            float* op = out + ((size_t)b * CO_real + n) * kHW + hw0 + hi * 8;
            v4f lo4 = __builtin_shufflevector(acc[nt], acc[nt], 0, 1, 2, 3);
            v4f hi4 = __builtin_shufflevector(acc[nt], acc[nt], 4, 5, 6, 7);
            lo4 += bv;
            hi4 += bv;
            if (accumulate) {
                const float* pp = probs + ((size_t)b * 4 + prob_ch) * kHW + hw0 + hi * 8;
                v4f pl = *(const v4f*)(pp);
                v4f ph = *(const v4f*)(pp + 4);
                v4f ol = *(v4f*)(op);
                v4f oh = *(v4f*)(op + 4);
                *(v4f*)(op)     = ol + pl * lo4;   // unique writer per element
                *(v4f*)(op + 4) = oh + ph * hi4;
            } else {
                *(v4f*)(op)     = lo4;
                *(v4f*)(op + 4) = hi4;
            }
        }
    }
}

// ---------------------------------------------------------------------------
__global__ void softmax4_kernel(const float* __restrict__ wlog,
                                float* __restrict__ probs)
{
    int i = blockIdx.x * blockDim.x + threadIdx.x;
    if (i >= kB * kHW) return;
    int b = i >> 14, hw = i & (kHW - 1);
    const float* p = wlog + (size_t)b * 4 * kHW + hw;
    float a0 = p[0], a1 = p[kHW], a2 = p[2 * kHW], a3 = p[3 * kHW];
    float mx = fmaxf(fmaxf(a0, a1), fmaxf(a2, a3));
    float e0 = expf(a0 - mx), e1 = expf(a1 - mx);
    float e2 = expf(a2 - mx), e3 = expf(a3 - mx);
    float inv = 1.f / (e0 + e1 + e2 + e3);
    float* q = probs + (size_t)b * 4 * kHW + hw;
    q[0] = e0 * inv; q[kHW] = e1 * inv; q[2 * kHW] = e2 * inv; q[3 * kHW] = e3 * inv;
}

// ---------------------------------------------------------------------------
extern "C" void kernel_launch(void* const* d_in, const int* in_sizes, int n_in,
                              void* d_out, int out_size, void* d_ws, size_t ws_size,
                              hipStream_t stream)
{
    const float* x        = (const float*)d_in[0];
    const float* bn_g     = (const float*)d_in[1];
    const float* bn_b     = (const float*)d_in[2];
    const float* bn_m     = (const float*)d_in[3];
    const float* bn_v     = (const float*)d_in[4];
    const float* offset_w = (const float*)d_in[5];   // [4,18,128,3,3]
    const float* offset_b = (const float*)d_in[6];   // [4,18]
    const float* dconv_w  = (const float*)d_in[7];   // [4,32,32,3,3]
    const float* dconv_b  = (const float*)d_in[8];   // [4,32]
    const float* wbn_g    = (const float*)d_in[9];
    const float* wbn_b    = (const float*)d_in[10];
    const float* wbn_m    = (const float*)d_in[11];
    const float* wbn_v    = (const float*)d_in[12];
    const float* woff_w   = (const float*)d_in[13];  // [18,128,3,3]
    const float* woff_b   = (const float*)d_in[14];  // [18]
    const float* wd_w     = (const float*)d_in[15];  // [4,128,3,3]
    const float* wd_b     = (const float*)d_in[16];  // [4]
    float* out = (float*)d_out;

    char* ws = (char*)d_ws;
    size_t cur = 0;
    auto alloc = [&](size_t bytes) {
        char* p = ws + cur;
        cur += (bytes + 255) & ~(size_t)255;
        return p;
    };
    size_t padded_halves = (size_t)kB * kHP * kWP * kC;
    _Float16* xa_nhwc = (_Float16*)alloc(padded_halves * 2);
    _Float16* y_nhwc  = (_Float16*)alloc(padded_halves * 2);
    float* woff_out   = (float*)alloc((size_t)kB * kOff * kHW * 4);
    float* wlog       = (float*)alloc((size_t)kB * 4 * kHW * 4);
    float* probs      = (float*)alloc((size_t)kB * 4 * kHW * 4);
    float* off_buf[4];
    for (int i = 0; i < 4; ++i) off_buf[i] = (float*)alloc((size_t)kB * kOff * kHW * 4);
    _Float16* woff_pack = (_Float16*)alloc((size_t)2 * 36 * 32 * 16 * 2);
    _Float16* wd_pack   = (_Float16*)alloc((size_t)1 * 36 * 32 * 16 * 2);
    _Float16* offw_pack[4];
    for (int i = 0; i < 4; ++i) offw_pack[i] = (_Float16*)alloc((size_t)2 * 36 * 32 * 16 * 2);
    _Float16* dcw_pack[4];
    for (int i = 0; i < 4; ++i) dcw_pack[i] = (_Float16*)alloc((size_t)2 * 9 * 32 * 16 * 2);

    // 1. clear padded activation buffers (halo must be zero), then BN+pack
    int n8 = (int)(padded_halves / 8);
    zero_h_kernel<<<(n8 + 255) / 256, 256, 0, stream>>>(xa_nhwc, n8);
    zero_h_kernel<<<(n8 + 255) / 256, 256, 0, stream>>>(y_nhwc, n8);
    bn_pack_kernel<<<(kB * kHW + 255) / 256, 256, 0, stream>>>(
        x, bn_g, bn_b, bn_m, bn_v, wbn_g, wbn_b, wbn_m, wbn_v, xa_nhwc, y_nhwc);

    // 2. pack all conv weights into WMMA B layout
    auto packLaunch = [&](const float* wsrc, _Float16* dst, int O, int Cin, int NT) {
        int total = NT * ((Cin * 9) / 32) * 32 * 16;
        pack_wmma_b_kernel<<<(total + 255) / 256, 256, 0, stream>>>(wsrc, dst, O, Cin, NT);
    };
    packLaunch(woff_w, woff_pack, kOff, 128, 2);
    packLaunch(wd_w, wd_pack, 4, 128, 1);
    for (int i = 0; i < 4; ++i)
        packLaunch(offset_w + (size_t)i * kOff * 128 * 9, offw_pack[i], kOff, 128, 2);
    for (int i = 0; i < 4; ++i)
        packLaunch(dconv_w + (size_t)i * 32 * 32 * 9, dcw_pack[i], 32, 32, 2);

    int tiles32 = kB * kHW / 32;   // conv: 32-pixel tiles
    int tiles16 = kB * kHW / 16;   // deform: 16-pixel tiles

    // 3. weighting path: conv(y) -> offsets, deform(y) -> wlog, softmax
    conv3x3_wmma_kernel<128, 2><<<tiles32, 32, 0, stream>>>(
        y_nhwc, woff_pack, woff_b, woff_out, kOff);
    deform_wmma_kernel<128, 1><<<tiles16, 32, 0, stream>>>(
        y_nhwc, 0, woff_out, wd_pack, wd_b, wlog, 4, nullptr, 0, 0);
    softmax4_kernel<<<(kB * kHW + 255) / 256, 256, 0, stream>>>(wlog, probs);

    // 4. output accumulation over the 4 deformable branches
    zero_kernel<<<(out_size + 255) / 256, 256, 0, stream>>>(out, out_size);
    for (int i = 0; i < 4; ++i) {
        conv3x3_wmma_kernel<128, 2><<<tiles32, 32, 0, stream>>>(
            xa_nhwc, offw_pack[i], offset_b + i * kOff, off_buf[i], kOff);
        deform_wmma_kernel<32, 2><<<tiles16, 32, 0, stream>>>(
            xa_nhwc, i * 32, off_buf[i], dcw_pack[i], dconv_b + i * 32,
            out, 32, probs, i, 1);
    }
}
